// DenoiserBlock_55688545960162
// MI455X (gfx1250) — compile-verified, matrix operand order checked by reading, not run on hardware
//
#include <hip/hip_runtime.h>

// ---------------- problem constants ----------------
#define B_      4
#define S_      1024
#define E_      512
#define INNER_  1024
#define NH_     4
#define KS_     4
#define DH_     256           // INNER_/NH_
#define MROWS   (B_*S_)       // 4096
#define UPCOLS  (2*INNER_)    // 2048

// ---------------- vector types ----------------
typedef __attribute__((ext_vector_type(16))) __bf16          bf16x16;
typedef __attribute__((ext_vector_type(8)))  float           f32x8;
typedef __attribute__((ext_vector_type(4)))  float           f32x4;
typedef __attribute__((ext_vector_type(8)))  unsigned short  u16x8;
typedef __attribute__((ext_vector_type(4)))  unsigned int    u32x4t;
typedef __attribute__((ext_vector_type(4)))  int             i32x4t;
typedef __attribute__((ext_vector_type(8)))  int             i32x8t;

union FragAB { bf16x16 v; u16x8 h[2]; unsigned short u[16]; };

__device__ __forceinline__ unsigned short f2bf(float f) {
  union { float f; unsigned u; } x; x.f = f;
  unsigned r = x.u + 0x7FFFu + ((x.u >> 16) & 1u);
  return (unsigned short)(r >> 16);
}
__device__ __forceinline__ float bf2f(unsigned short h) {
  union { unsigned u; float f; } x; x.u = ((unsigned)h) << 16; return x.f;
}

// 16-bit A/B fragment K-mapping (ISA 7.12.2): per lane the 16 elements are two
// contiguous 8-element runs: K = half*8 + {0..7} and K = 16 + half*8 + {0..7}.
// => a fragment is two b128 loads when the operand is K-contiguous in LDS.
__device__ __forceinline__ void load_frag(FragAB& f, const unsigned short* base, int half) {
  f.h[0] = *(const u16x8*)(base + half * 8);
  f.h[1] = *(const u16x8*)(base + 16 + half * 8);
}

__device__ __forceinline__ f32x8 wmma_bf16(bf16x16 a, bf16x16 b, f32x8 c) {
  return __builtin_amdgcn_wmma_f32_16x16x32_bf16(false, a, false, b, (short)0, c,
                                                 false, false);
}

// ---------------- Tensor Data Mover (gfx1250 TDM) ----------------
#if defined(__has_builtin)
#  if __has_builtin(__builtin_amdgcn_tensor_load_to_lds)
#    define HAVE_TDM 1
#  endif
#endif
#ifndef HAVE_TDM
#  define HAVE_TDM 0
#endif

#if HAVE_TDM
// 2D tile load: tile_d0 elements/row (2-byte units), tile_d1 rows,
// row stride `stride0` elements, into LDS byte offset `lds_off`.
__device__ __forceinline__ void tdm_load_2d(unsigned lds_off, const void* gptr,
                                            unsigned tile_d0, unsigned tile_d1,
                                            unsigned stride0, unsigned tensor_d1) {
  unsigned long long ga = (unsigned long long)(size_t)gptr;
  u32x4t g0;
  g0[0] = 1u;                                     // count=1, user descriptor
  g0[1] = lds_off;                                // lds_addr
  g0[2] = (unsigned)(ga & 0xFFFFFFFFu);           // global_addr[31:0]
  g0[3] = (unsigned)((ga >> 32) & 0x01FFFFFFu) | (2u << 30);  // addr[56:32], type=2
  i32x8t g1;
  g1[0] = (int)(1u << 16);                        // data_size=1 (2 bytes)
  g1[1] = (int)((tile_d0 & 0xFFFFu) << 16);       // tensor_dim0[15:0] @ bit48
  g1[2] = (int)((tile_d0 >> 16) | ((tensor_d1 & 0xFFFFu) << 16)); // dim0 hi | dim1 lo
  g1[3] = (int)((tensor_d1 >> 16) | ((tile_d0 & 0xFFFFu) << 16)); // dim1 hi | tile_dim0
  g1[4] = (int)(tile_d1 & 0xFFFFu);               // tile_dim1 (tile_dim2=0 -> 2D)
  g1[5] = (int)stride0;                           // tensor_dim0_stride[31:0]
  g1[6] = 0; g1[7] = 0;
  i32x4t gz = {0, 0, 0, 0};
#if __clang_major__ >= 23
  i32x8t gz8 = {0, 0, 0, 0, 0, 0, 0, 0};
  __builtin_amdgcn_tensor_load_to_lds(g0, g1, gz, gz, gz8, 0);
#else
  __builtin_amdgcn_tensor_load_to_lds(g0, g1, gz, gz, 0);
#endif
}
#endif

// ---------------- K1: layernorm -> bf16 ----------------
__global__ __launch_bounds__(256) void ln_kernel(
    const float* __restrict__ x, const float* __restrict__ w,
    const float* __restrict__ b, unsigned short* __restrict__ xn) {
  int row = blockIdx.x, tid = threadIdx.x;
  __shared__ float red[256];
  const float* xr = x + (size_t)row * E_;
  float s = 0.f;
  for (int i = tid; i < E_; i += 256) s += xr[i];
  red[tid] = s; __syncthreads();
  for (int st = 128; st > 0; st >>= 1) { if (tid < st) red[tid] += red[tid + st]; __syncthreads(); }
  float mu = red[0] * (1.f / E_); __syncthreads();
  float v = 0.f;
  for (int i = tid; i < E_; i += 256) { float d = xr[i] - mu; v += d * d; }
  red[tid] = v; __syncthreads();
  for (int st = 128; st > 0; st >>= 1) { if (tid < st) red[tid] += red[tid + st]; __syncthreads(); }
  float rstd = rsqrtf(red[0] * (1.f / E_) + 1e-5f); __syncthreads();
  for (int i = tid; i < E_; i += 256)
    xn[(size_t)row * E_ + i] = f2bf((xr[i] - mu) * rstd * w[i] + b[i]);
}

// ---------------- K2: f32 -> bf16 ----------------
__global__ __launch_bounds__(256) void cvt_kernel(
    const float* __restrict__ s, unsigned short* __restrict__ d, int n) {
  int i = blockIdx.x * blockDim.x + threadIdx.x;
  if (i < n) d[i] = f2bf(s[i]);
}

// ---------------- K3: bf16 WMMA GEMM, TDM-staged A, transposed-B LDS ------
#define TILE_M 64
#define TILE_N 64
#define TILE_K 32
__global__ __launch_bounds__(128) void gemm_bf16_wmma(
    const unsigned short* __restrict__ A, const unsigned short* __restrict__ Bm,
    float* __restrict__ C, const float* __restrict__ addend,
    int M, int N, int K) {
  // single arena => As is at LDS offset 0 (needed by the TDM descriptor)
  __shared__ __align__(16) unsigned char gsm[TILE_M*TILE_K*2 + TILE_N*TILE_K*2];
  unsigned short* As  = (unsigned short*)gsm;                       // [m][k] 64x32
  unsigned short* BsT = (unsigned short*)(gsm + TILE_M*TILE_K*2);   // [n][k] 64x32
  const int tid  = threadIdx.x;
  const int lane = tid & 31, wave = tid >> 5;
  const int wy = wave >> 1, wx = wave & 1;
  const int m0 = blockIdx.y * TILE_M, n0 = blockIdx.x * TILE_N;
  const int half = lane >> 4, l15 = lane & 15;

  f32x8 acc[2][2];
  for (int i = 0; i < 2; ++i) for (int j = 0; j < 2; ++j)
    for (int r = 0; r < 8; ++r) acc[i][j][r] = 0.f;

  for (int k0 = 0; k0 < K; k0 += TILE_K) {
    __syncthreads();     // previous tile fully consumed
#if HAVE_TDM
    if (wave == 0)       // async DMA: 64x32 bf16 tile, row stride K
      tdm_load_2d(0u, &A[(size_t)m0 * K + k0], TILE_K, TILE_M, (unsigned)K,
                  (unsigned)M);
#else
    for (int i = 0; i < 2; ++i) {             // 256 chunks of 8 ushorts
      int c = tid + i * 128, r = c >> 2, cc = (c & 3) * 8;
      *(u16x8*)&As[r * TILE_K + cc] =
          *(const u16x8*)&A[(size_t)(m0 + r) * K + k0 + cc];
    }
#endif
    // stage B transposed: read row-major rows (coalesced b128), scatter to [n][k]
    for (int i = 0; i < 2; ++i) {
      int c = tid + i * 128, kr = c >> 3, nc = (c & 7) * 8;
      u16x8 tv = *(const u16x8*)&Bm[(size_t)(k0 + kr) * N + n0 + nc];
      #pragma unroll
      for (int e = 0; e < 8; ++e) BsT[(nc + e) * TILE_K + kr] = tv[e];
    }
    if (k0 + TILE_K < K)
      __builtin_prefetch(&Bm[(size_t)(k0 + TILE_K + (tid >> 2)) * N + n0], 0, 1);
#if HAVE_TDM
    if (wave == 0) __builtin_amdgcn_s_wait_tensorcnt(0);
#endif
    __syncthreads();

    FragAB afr[2], bfr[2];
    #pragma unroll
    for (int i = 0; i < 2; ++i)
      load_frag(afr[i], &As[(wy * 32 + i * 16 + l15) * TILE_K], half);
    #pragma unroll
    for (int j = 0; j < 2; ++j)
      load_frag(bfr[j], &BsT[(wx * 32 + j * 16 + l15) * TILE_K], half);
    #pragma unroll
    for (int i = 0; i < 2; ++i)
      #pragma unroll
      for (int j = 0; j < 2; ++j)
        acc[i][j] = wmma_bf16(afr[i].v, bfr[j].v, acc[i][j]);
  }
  for (int i = 0; i < 2; ++i) for (int j = 0; j < 2; ++j) {
    #pragma unroll
    for (int r = 0; r < 8; ++r) {
      int row = m0 + wy * 32 + i * 16 + r + 8 * half;
      int col = n0 + wx * 32 + j * 16 + l15;
      size_t off = (size_t)row * N + col;
      float v = acc[i][j][r];
      if (addend) v += addend[off];
      C[off] = v;
    }
  }
}

// ---------------- K4: depthwise conv + swish ----------------
__global__ __launch_bounds__(256) void conv_swish_kernel(
    const float* __restrict__ up, const float* __restrict__ w,
    const float* __restrict__ bias, float* __restrict__ out, int rev) {
  int idx = blockIdx.x * blockDim.x + threadIdx.x;
  int ch = idx % INNER_;
  int s  = (idx / INNER_) % S_;
  int b  = idx / (INNER_ * S_);
  float acc = bias[ch];
  #pragma unroll
  for (int k = 0; k < KS_; ++k) {
    int t = s + k - 1;
    if (t >= 0 && t < S_) {
      int so = rev ? (S_ - 1 - t) : t;
      acc += up[(size_t)(b * S_ + so) * UPCOLS + ch] * w[ch * KS_ + k];
    }
  }
  out[idx] = acc / (1.f + __expf(-acc));
}

// ---------------- K5: headwise q/k/v (bf16 out) ----------------
__global__ __launch_bounds__(256) void headwise_kernel(
    const float* __restrict__ c, const float* __restrict__ up,
    const float* __restrict__ Wq, const float* __restrict__ Wk,
    const float* __restrict__ Wv, unsigned short* __restrict__ q,
    unsigned short* __restrict__ k, unsigned short* __restrict__ v, int rev) {
  int idx = blockIdx.x * blockDim.x + threadIdx.x;
  int i = idx % INNER_;
  int s = (idx / INNER_) % S_;
  int b = idx / (INNER_ * S_);
  int nb = i >> 2, oo = i & 3;
  const float* cr = c + (size_t)(b * S_ + s) * INNER_ + nb * 4;
  int so = rev ? (S_ - 1 - s) : s;
  const float* xr = up + (size_t)(b * S_ + so) * UPCOLS + nb * 4;
  float aq = 0.f, ak = 0.f, av = 0.f;
  #pragma unroll
  for (int ii = 0; ii < 4; ++ii) {
    int wi = nb * 16 + oo * 4 + ii;
    aq += cr[ii] * Wq[wi]; ak += cr[ii] * Wk[wi]; av += xr[ii] * Wv[wi];
  }
  q[idx] = f2bf(aq); k[idx] = f2bf(ak); v[idx] = f2bf(av);
}

// ---------------- K6: gate pre-activations ----------------
__global__ __launch_bounds__(256) void gates_kernel(
    const unsigned short* __restrict__ q, const unsigned short* __restrict__ k,
    const unsigned short* __restrict__ v, const float* __restrict__ Wi,
    const float* __restrict__ bi, const float* __restrict__ Wf,
    const float* __restrict__ bfv, float* __restrict__ ipre,
    float* __restrict__ fpre) {
  int bs = blockIdx.x;
  int b = bs / S_, s = bs % S_;
  int tid = threadIdx.x;
  __shared__ float rI[NH_][256];
  __shared__ float rF[NH_][256];
  float aI[NH_] = {0,0,0,0}, aF[NH_] = {0,0,0,0};
  const unsigned short* bufs[3] = {q, k, v};
  for (int t = tid; t < 3 * INNER_; t += 256) {
    float g = bf2f(bufs[t / INNER_][(size_t)bs * INNER_ + (t % INNER_)]);
    #pragma unroll
    for (int h = 0; h < NH_; ++h) { aI[h] += g * Wi[t*NH_+h]; aF[h] += g * Wf[t*NH_+h]; }
  }
  for (int h = 0; h < NH_; ++h) { rI[h][tid] = aI[h]; rF[h][tid] = aF[h]; }
  __syncthreads();
  for (int st = 128; st > 0; st >>= 1) {
    if (tid < st)
      for (int h = 0; h < NH_; ++h) { rI[h][tid] += rI[h][tid+st]; rF[h][tid] += rF[h][tid+st]; }
    __syncthreads();
  }
  if (tid < NH_) {
    ipre[((size_t)b * NH_ + tid) * S_ + s] = rI[tid][0] + bi[tid];
    fpre[((size_t)b * NH_ + tid) * S_ + s] = rF[tid][0] + bfv[tid];
  }
}

// ---------------- K7: cumsum(log-sigmoid) + prefix max ----------------
__global__ void cum_kernel(const float* __restrict__ ipre,
                           const float* __restrict__ fpre,
                           float* __restrict__ cum, float* __restrict__ maxD) {
  int bh = blockIdx.x;
  if (threadIdx.x != 0) return;
  const float* fp = fpre + (size_t)bh * S_;
  const float* ip = ipre + (size_t)bh * S_;
  float c = 0.f, rm = -1e30f;
  for (int s = 0; s < S_; ++s) {
    float xv = fp[s];
    float ls = (xv >= 0.f) ? -log1pf(__expf(-xv)) : (xv - log1pf(__expf(xv)));
    c += ls;
    cum[(size_t)bh * S_ + s] = c;
    rm = fmaxf(rm, ip[s] - c);
    maxD[(size_t)bh * S_ + s] = c + rm;
  }
}

// ---------------- K8: streaming decay-attention (WMMA + TDM) --------------
#define ATT_WAVES 2
// arena layout (byte offsets): ks @0 (16K), vsT @16K (16K), qs @32K (16K),
// sc @49152 (4K), aux @53248
#define KS_OFF   0
#define VST_OFF  16384
#define QS_OFF   32768
#define SC_OFF   49152
#define AUX_OFF  53248
__global__ __launch_bounds__(64) void attention_kernel(
    const unsigned short* __restrict__ q, const unsigned short* __restrict__ k,
    const unsigned short* __restrict__ v, const float* __restrict__ ipre,
    const float* __restrict__ cum, const float* __restrict__ maxD,
    float* __restrict__ hcell) {
  __shared__ __align__(16) unsigned char smem[53632];
  unsigned short* ks  = (unsigned short*)(smem + KS_OFF);   // [t][d] 32x256
  unsigned short* vsT = (unsigned short*)(smem + VST_OFF);  // [d][t] 256x32
  unsigned short* qs  = (unsigned short*)(smem + QS_OFF);   // per wave 16x256
  float* sc  = (float*)(smem + SC_OFF);                     // per wave 16x32
  float* aux = (float*)(smem + AUX_OFF);                    // per wave 48

  int tid = threadIdx.x;
  int lane = tid & 31, wv = tid >> 5;
  int gw = blockIdx.x * ATT_WAVES + wv;
  int qt = gw & 63, bh = gw >> 6;
  int b = bh >> 2, h = bh & 3;
  int r0 = qt * 16;
  int r0blk = ((blockIdx.x * ATT_WAVES) & 63) * 16;
  int blkmax = r0blk + 16 * ATT_WAVES - 1;      // uniform loop bound for block
  int half = lane >> 4, l15 = lane & 15;

  const unsigned short* qbase = q + (size_t)(b * S_) * INNER_ + h * DH_;
  const unsigned short* kbase = k + (size_t)(b * S_) * INNER_ + h * DH_;
  const unsigned short* vbase = v + (size_t)(b * S_) * INNER_ + h * DH_;
  const float* cumA = cum  + (size_t)bh * S_;
  const float* ipA  = ipre + (size_t)bh * S_;

  unsigned short* qsw = qs + wv * 16 * DH_;
  float* scw  = sc  + wv * 16 * 32;
  float* auxw = aux + wv * 48;

  // stage this wave's 16x256 q tile (b128 in / b128 out)
  for (int c = lane; c < (16 * DH_) / 8; c += 32) {
    int rr = c >> 5, dd = (c & 31) * 8;
    *(u16x8*)&qsw[rr * DH_ + dd] =
        *(const u16x8*)&qbase[(size_t)(r0 + rr) * INNER_ + dd];
  }
  if (lane < 16) {
    auxw[lane]      = cumA[r0 + lane];
    auxw[16 + lane] = maxD[(size_t)bh * S_ + r0 + lane];
  }

  float rowsum = 0.f;
  f32x8 hacc[16];
  for (int t = 0; t < 16; ++t) for (int r = 0; r < 8; ++r) hacc[t][r] = 0.f;

  const float scale = 0.0625f;                  // DH^-0.5
  int smax = r0 + 15;
  for (int t0 = 0; t0 <= blkmax; t0 += 32) {
    __syncthreads();                            // previous K/V tiles consumed
#if HAVE_TDM
    if (wv == 0)                                // async DMA K tile 32x256
      tdm_load_2d(KS_OFF, kbase + (size_t)t0 * INNER_, DH_, 32, INNER_, S_);
#endif
    // cooperative V transpose stage (and K fallback) with all 64 threads
    for (int c = tid; c < (32 * DH_) / 8; c += 64) {
      int t = c >> 5, dd = (c & 31) * 8;
#if !HAVE_TDM
      *(u16x8*)&ks[t * DH_ + dd] =
          *(const u16x8*)&kbase[(size_t)(t0 + t) * INNER_ + dd];
#endif
      u16x8 vv = *(const u16x8*)&vbase[(size_t)(t0 + t) * INNER_ + dd];
      #pragma unroll
      for (int e = 0; e < 8; ++e) vsT[(dd + e) * 32 + t] = vv[e];
    }
#if HAVE_TDM
    if (wv == 0) __builtin_amdgcn_s_wait_tensorcnt(0);
#endif
    __syncthreads();

    if (t0 <= smax) {
      // ---- score panel: two 16x16 QK^T WMMA tiles ----
      for (int hh = 0; hh < 2; ++hh) {
        int tc = t0 + hh * 16;
        if (tc <= smax) {
          f32x8 acc; for (int r = 0; r < 8; ++r) acc[r] = 0.f;
          for (int kk = 0; kk < DH_; kk += 32) {
            FragAB af, bf_;
            load_frag(af,  &qsw[l15 * DH_ + kk], half);
            load_frag(bf_, &ks[(hh * 16 + l15) * DH_ + kk], half);
            acc = wmma_bf16(af.v, bf_.v, acc);
          }
          #pragma unroll
          for (int r = 0; r < 8; ++r) {
            int m = r + 8 * half;
            int srow = r0 + m, tcol = tc + l15;
            float cmv = 0.f;
            if (tcol <= srow) {
              float ld = auxw[m] - cumA[tcol] + ipA[tcol] - auxw[16 + m];
              cmv = acc[r] * scale * __expf(ld);
            }
            scw[m * 32 + hh * 16 + l15] = cmv;
          }
        } else {
          #pragma unroll
          for (int r = 0; r < 8; ++r)
            scw[(r + 8 * half) * 32 + hh * 16 + l15] = 0.f;
        }
      }
      if (lane < 16) {                          // per-wave LDS is in-order
        float s = 0.f;
        #pragma unroll
        for (int c = 0; c < 32; ++c) s += scw[lane * 32 + c];
        rowsum += s;
      }
      // ---- PV: scores(16x32) @ V(32x16 per DH N-tile) ----
      FragAB af;
      {
        f32x4 a0 = *(const f32x4*)&scw[l15 * 32 + half * 8];
        f32x4 a1 = *(const f32x4*)&scw[l15 * 32 + half * 8 + 4];
        f32x4 a2 = *(const f32x4*)&scw[l15 * 32 + 16 + half * 8];
        f32x4 a3 = *(const f32x4*)&scw[l15 * 32 + 16 + half * 8 + 4];
        #pragma unroll
        for (int e = 0; e < 4; ++e) {
          af.u[e]      = f2bf(a0[e]);  af.u[4 + e]  = f2bf(a1[e]);
          af.u[8 + e]  = f2bf(a2[e]);  af.u[12 + e] = f2bf(a3[e]);
        }
      }
      for (int nt = 0; nt < 16; ++nt) {
        FragAB bf_;
        load_frag(bf_, &vsT[(nt * 16 + l15) * 32], half);
        hacc[nt] = wmma_bf16(af.v, bf_.v, hacc[nt]);
      }
    }
  }
  if (lane < 16) {
    float nrm = fmaxf(fabsf(rowsum), __expf(-auxw[16 + lane]));
    auxw[32 + lane] = 1.f / (nrm + 1e-6f);
  }
  float* hout = hcell + (size_t)(b * S_) * INNER_ + h * DH_;
  for (int nt = 0; nt < 16; ++nt) {
    #pragma unroll
    for (int r = 0; r < 8; ++r) {
      int m = r + 8 * half;
      hout[(size_t)(r0 + m) * INNER_ + nt * 16 + l15] = hacc[nt][r] * auxw[32 + m];
    }
  }
}

// ---------------- K9: per-head groupnorm * on_w + skip * c ----------------
__global__ __launch_bounds__(256) void gnorm_kernel(
    const float* __restrict__ hcell, const float* __restrict__ c,
    const float* __restrict__ on_w, const float* __restrict__ skip,
    float* __restrict__ hdir) {
  int bsh = blockIdx.x;
  int h = bsh % NH_, bs = bsh / NH_;
  int tid = threadIdx.x;
  __shared__ float red[256];
  float val = hcell[(size_t)bs * INNER_ + h * DH_ + tid];
  red[tid] = val; __syncthreads();
  for (int st = 128; st > 0; st >>= 1) { if (tid < st) red[tid] += red[tid+st]; __syncthreads(); }
  float mu = red[0] * (1.f / DH_); __syncthreads();
  float d = val - mu; red[tid] = d * d; __syncthreads();
  for (int st = 128; st > 0; st >>= 1) { if (tid < st) red[tid] += red[tid+st]; __syncthreads(); }
  float rstd = rsqrtf(red[0] * (1.f / DH_) + 1e-5f);
  int i = h * DH_ + tid;
  hdir[(size_t)bs * INNER_ + i] =
      (val - mu) * rstd * on_w[i] + skip[i] * c[(size_t)bs * INNER_ + i];
}

// ---------------- K10: combine + swish(z) gate -> bf16 ----------------
__global__ __launch_bounds__(256) void combine_kernel(
    const float* __restrict__ hf, const float* __restrict__ hr,
    const float* __restrict__ up, unsigned short* __restrict__ hb) {
  int idx = blockIdx.x * blockDim.x + threadIdx.x;
  int i = idx % INNER_;
  int s = (idx / INNER_) % S_;
  int b = idx / (INNER_ * S_);
  float z = up[(size_t)(b * S_ + s) * UPCOLS + INNER_ + i];
  float sw = z / (1.f + __expf(-z));
  float hv = hf[idx] + hr[(size_t)(b * S_ + (S_ - 1 - s)) * INNER_ + i];
  hb[idx] = f2bf(hv * sw);
}

// ---------------- host orchestration ----------------
extern "C" void kernel_launch(void* const* d_in, const int* in_sizes, int n_in,
                              void* d_out, int out_size, void* d_ws, size_t ws_size,
                              hipStream_t stream) {
  (void)in_sizes; (void)n_in; (void)out_size; (void)ws_size;
  const float* x    = (const float*)d_in[0];
  const float* ln_w = (const float*)d_in[1];
  const float* ln_b = (const float*)d_in[2];
  const float* W_up = (const float*)d_in[3];
  const float* W_dn = (const float*)d_in[4];
  struct DirIn { const float *cw, *cb, *Wq, *Wk, *Wv, *Wi, *bi, *Wf, *bf, *skip, *onw; };
  DirIn din[2];
  for (int d = 0; d < 2; ++d) {
    int base = 5 + d * 11;
    din[d].cw  = (const float*)d_in[base + 0];
    din[d].cb  = (const float*)d_in[base + 1];
    din[d].Wq  = (const float*)d_in[base + 2];
    din[d].Wk  = (const float*)d_in[base + 3];
    din[d].Wv  = (const float*)d_in[base + 4];
    din[d].Wi  = (const float*)d_in[base + 5];
    din[d].bi  = (const float*)d_in[base + 6];
    din[d].Wf  = (const float*)d_in[base + 7];
    din[d].bf  = (const float*)d_in[base + 8];
    din[d].skip= (const float*)d_in[base + 9];
    din[d].onw = (const float*)d_in[base + 10];
  }
  float* out = (float*)d_out;

  char* ws = (char*)d_ws; size_t o = 0;
  auto alloc = [&](size_t bytes) -> char* {
    char* p = ws + o; o += (bytes + 255) & ~(size_t)255; return p;
  };
  unsigned short* xn_bf  = (unsigned short*)alloc((size_t)MROWS * E_ * 2);
  unsigned short* wup_bf = (unsigned short*)alloc((size_t)E_ * UPCOLS * 2);
  unsigned short* wdn_bf = (unsigned short*)alloc((size_t)INNER_ * E_ * 2);
  float* up   = (float*)alloc((size_t)MROWS * UPCOLS * 4);
  float* cbuf = (float*)alloc((size_t)MROWS * INNER_ * 4);
  unsigned short* qb = (unsigned short*)alloc((size_t)MROWS * INNER_ * 2);
  unsigned short* kb = (unsigned short*)alloc((size_t)MROWS * INNER_ * 2);
  unsigned short* vb = (unsigned short*)alloc((size_t)MROWS * INNER_ * 2);
  float* ipre = (float*)alloc((size_t)B_ * NH_ * S_ * 4);
  float* fpre = (float*)alloc((size_t)B_ * NH_ * S_ * 4);
  float* cumb = (float*)alloc((size_t)B_ * NH_ * S_ * 4);
  float* maxD = (float*)alloc((size_t)B_ * NH_ * S_ * 4);
  float* hcell = (float*)alloc((size_t)MROWS * INNER_ * 4);
  float* hdir0 = (float*)alloc((size_t)MROWS * INNER_ * 4);
  float* hdir1 = (float*)alloc((size_t)MROWS * INNER_ * 4);
  unsigned short* hb = (unsigned short*)alloc((size_t)MROWS * INNER_ * 2);
  float* hdirs[2] = {hdir0, hdir1};

  ln_kernel<<<MROWS, 256, 0, stream>>>(x, ln_w, ln_b, xn_bf);
  {
    int n1 = E_ * UPCOLS, n2 = INNER_ * E_;
    cvt_kernel<<<(n1 + 255) / 256, 256, 0, stream>>>(W_up, wup_bf, n1);
    cvt_kernel<<<(n2 + 255) / 256, 256, 0, stream>>>(W_dn, wdn_bf, n2);
  }
  gemm_bf16_wmma<<<dim3(UPCOLS / TILE_N, MROWS / TILE_M), 128, 0, stream>>>(
      xn_bf, wup_bf, up, nullptr, MROWS, UPCOLS, E_);

  const int ELEM_BLOCKS = (MROWS * INNER_) / 256;
  for (int d = 0; d < 2; ++d) {
    int rev = d;
    conv_swish_kernel<<<ELEM_BLOCKS, 256, 0, stream>>>(up, din[d].cw, din[d].cb, cbuf, rev);
    headwise_kernel<<<ELEM_BLOCKS, 256, 0, stream>>>(cbuf, up, din[d].Wq, din[d].Wk,
                                                     din[d].Wv, qb, kb, vb, rev);
    gates_kernel<<<MROWS, 256, 0, stream>>>(qb, kb, vb, din[d].Wi, din[d].bi,
                                            din[d].Wf, din[d].bf, ipre, fpre);
    cum_kernel<<<B_ * NH_, 32, 0, stream>>>(ipre, fpre, cumb, maxD);
    attention_kernel<<<(B_ * NH_ * (S_ / 16)) / ATT_WAVES, 32 * ATT_WAVES, 0, stream>>>(
        qb, kb, vb, ipre, cumb, maxD, hcell);
    gnorm_kernel<<<MROWS * NH_, 256, 0, stream>>>(hcell, cbuf, din[d].onw,
                                                  din[d].skip, hdirs[d]);
  }
  combine_kernel<<<ELEM_BLOCKS, 256, 0, stream>>>(hdir0, hdir1, up, hb);
  gemm_bf16_wmma<<<dim3(E_ / TILE_N, MROWS / TILE_M), 128, 0, stream>>>(
      hb, wdn_bf, out, x, MROWS, E_, INNER_);
}
